// W8A8B8O8Linear_541165879313
// MI455X (gfx1250) — 8-bit matrix operand order repaired and checked by reading, compiled, not run on hardware
//
#include <hip/hip_runtime.h>
#include <stdint.h>


typedef __attribute__((ext_vector_type(8))) int v8i;
typedef __attribute__((ext_vector_type(4))) int v4i;
typedef __attribute__((ext_vector_type(2))) int v2i;

__device__ __forceinline__ v8i load_afrag(const int8_t* __restrict__ p) {
    v2i w0 = *(const v2i*)(p + 0);
    v2i w1 = *(const v2i*)(p + 16);
    v2i w2 = *(const v2i*)(p + 32);
    v2i w3 = *(const v2i*)(p + 48);
    v8i a;
    a[0] = w0[0]; a[1] = w0[1];
    a[2] = w1[0]; a[3] = w1[1];
    a[4] = w2[0]; a[5] = w2[1];
    a[6] = w3[0]; a[7] = w3[1];
    return a;
}

__device__ __forceinline__ v8i load_bfrag(const int8_t* __restrict__ p) {
    v2i w0 = *(const v2i*)(p + 0);
    v2i w1 = *(const v2i*)(p + 16);
    v2i w2 = *(const v2i*)(p + 32);
    v2i w3 = *(const v2i*)(p + 48);
    v8i b;
    b[0] = w0[0]; b[1] = w0[1]; b[2] = w1[0]; b[3] = w1[1];
    b[4] = w2[0]; b[5] = w2[1]; b[6] = w3[0]; b[7] = w3[1];
    return b;
}

__global__ __launch_bounds__(256)
void w8a8b8o8_linear_kernel(const int8_t* __restrict__ x,
                            const int8_t* __restrict__ w,
                            const int8_t* __restrict__ bias,
                            const float*  __restrict__ alpha_p,
                            const float*  __restrict__ beta_p,
                            int8_t* __restrict__ out,
                            int K, int OUT) {
    const int lane = threadIdx.x & 31;
    const int wave = threadIdx.x >> 5;
    const int wm   = wave & 1;
    const int wn   = wave >> 1;
    const int l15  = lane & 15;
    const int sel  = lane >> 4;

    const int m_base = blockIdx.y * 64  + wm * 32;
    const int n_base = blockIdx.x * 256 + wn * 64;

    const int8_t* ap0 = x + (size_t)(m_base +  0 + l15) * K + sel * 8;
    const int8_t* ap1 = x + (size_t)(m_base + 16 + l15) * K + sel * 8;
    const int8_t* bp0 = w + (size_t)(n_base +  0 + l15) * K + sel * 8;
    const int8_t* bp1 = w + (size_t)(n_base + 16 + l15) * K + sel * 8;
    const int8_t* bp2 = w + (size_t)(n_base + 32 + l15) * K + sel * 8;
    const int8_t* bp3 = w + (size_t)(n_base + 48 + l15) * K + sel * 8;

    v8i acc[2][4];
#pragma unroll
    for (int i = 0; i < 2; ++i)
#pragma unroll
        for (int j = 0; j < 4; ++j)
            acc[i][j] = (v8i){0, 0, 0, 0, 0, 0, 0, 0};

    for (int k = 0; k < K; k += 64) {
        v8i a0 = load_afrag(ap0); ap0 += 64;
        v8i a1 = load_afrag(ap1); ap1 += 64;
        v8i b0 = load_bfrag(bp0); bp0 += 64;
        v8i b1 = load_bfrag(bp1); bp1 += 64;
        v8i b2 = load_bfrag(bp2); bp2 += 64;
        v8i b3 = load_bfrag(bp3); bp3 += 64;

        acc[0][0] = __builtin_amdgcn_wmma_i32_16x16x64_iu8(true, a0, true, b0, acc[0][0], false, false);
        acc[0][1] = __builtin_amdgcn_wmma_i32_16x16x64_iu8(true, a0, true, b1, acc[0][1], false, false);
        acc[0][2] = __builtin_amdgcn_wmma_i32_16x16x64_iu8(true, a0, true, b2, acc[0][2], false, false);
        acc[0][3] = __builtin_amdgcn_wmma_i32_16x16x64_iu8(true, a0, true, b3, acc[0][3], false, false);
        acc[1][0] = __builtin_amdgcn_wmma_i32_16x16x64_iu8(true, a1, true, b0, acc[1][0], false, false);
        acc[1][1] = __builtin_amdgcn_wmma_i32_16x16x64_iu8(true, a1, true, b1, acc[1][1], false, false);
        acc[1][2] = __builtin_amdgcn_wmma_i32_16x16x64_iu8(true, a1, true, b2, acc[1][2], false, false);
        acc[1][3] = __builtin_amdgcn_wmma_i32_16x16x64_iu8(true, a1, true, b3, acc[1][3], false, false);
    }

    const float alpha = *alpha_p;
    const float beta  = *beta_p;

#pragma unroll
    for (int nt = 0; nt < 4; ++nt) {
        const int col = n_base + nt * 16 + l15;
        const float bb = beta * (float)bias[col];
#pragma unroll
        for (int mt = 0; mt < 2; ++mt) {
            const int row0 = m_base + mt * 16 + sel * 8;
#pragma unroll
            for (int r = 0; r < 8; ++r) {
                float f = fmaf(alpha, (float)acc[mt][nt][r], bb);
                f = __builtin_rintf(f);
                f = fminf(fmaxf(f, -128.0f), 127.0f);
                out[(size_t)(row0 + r) * OUT + col] = (int8_t)(int)f;
            }
        }
    }
}

extern "C" void kernel_launch(void* const* d_in, const int* in_sizes, int n_in,
                              void* d_out, int out_size, void* d_ws, size_t ws_size,
                              hipStream_t stream) {
    const int8_t* x     = (const int8_t*)d_in[0];
    const int8_t* w     = (const int8_t*)d_in[1];
    const int8_t* bias  = (const int8_t*)d_in[2];
    const float*  alpha = (const float*)d_in[3];
    const float*  beta  = (const float*)d_in[4];
    int8_t*       out   = (int8_t*)d_out;

    const int OUT   = in_sizes[2];
    const int K     = in_sizes[1] / OUT;
    const int NROWS = in_sizes[0] / K;

    dim3 block(256);
    dim3 grid(OUT / 256, NROWS / 64);
    w8a8b8o8_linear_kernel<<<grid, block, 0, stream>>>(x, w, bias, alpha, beta,
                                                       out, K, OUT);
}
